// PointNet2Encoder_35450660062077
// MI455X (gfx1250) — matrix operand order repaired and checked by reading, not compiled, not run on hardware
//
#include <hip/hip_runtime.h>
#include <hip/hip_bf16.h>
#include <math.h>

// ---------------------------------------------------------------------------
// PointNet++ encoder for MI455X (gfx1250, wave32, WMMA).
// All activations are point-major (row = point, K channels contiguous) so both
// GEMM operands are K-contiguous: staging needs no transpose and uses the
// CDNA5 async global->LDS path (ASYNCcnt) when available.
// ---------------------------------------------------------------------------

typedef __attribute__((ext_vector_type(16))) _Float16 v16h;
typedef __attribute__((ext_vector_type(8)))  float    v8f;
typedef __attribute__((ext_vector_type(4)))  int      v4i;

#define NB   128
#define NPT  8192
#define EPSF 1e-5f

#if defined(__has_builtin)
#if __has_builtin(__builtin_amdgcn_global_load_async_to_lds_b128) && \
    __has_builtin(__builtin_amdgcn_s_wait_asynccnt)
#define USE_ASYNC_LDS 1
#endif
#endif
#ifndef USE_ASYNC_LDS
#define USE_ASYNC_LDS 0
#endif

#if USE_ASYNC_LDS
__device__ __forceinline__ void async_copy_b128(const _Float16* g, _Float16* l) {
    __builtin_amdgcn_global_load_async_to_lds_b128((v4i*)g, (v4i*)l, 0, 0);
}
#endif

// ------------------------------- utility -----------------------------------

__global__ void k_f32_to_f16(const float* __restrict__ in, _Float16* __restrict__ out, int n) {
    int i = blockIdx.x * 256 + threadIdx.x;
    if (i < n) out[i] = (_Float16)in[i];
}

// --------------------------- SA1 grouping ----------------------------------
// One block per batch. dist[] persists in LDS; xyz reads hit L2 (96KB/batch,
// 192MB L2 holds the full cloud). Output grouped1 is point-major (b*2048+e, 9).
__global__ __launch_bounds__(256) void k_sa1_group(
    const float* __restrict__ pc,       // (B, 8192, 6)
    float* __restrict__ new_xyz,        // (B, 32, 3)
    _Float16* __restrict__ grouped)     // (B*2048, 9)
{
    __shared__ float dist[NPT];
    __shared__ float redv[256];
    __shared__ int   redi[256];
    __shared__ float cenx[32], ceny[32], cenz[32];
    __shared__ int   idxl[32 * 64];
    __shared__ int   cntArr[32];

    const int b   = blockIdx.x;
    const int tid = threadIdx.x;
    const int lane = tid & 31, w = tid >> 5;
    const float* pcb = pc + (size_t)b * NPT * 6;

    for (int i = tid; i < NPT; i += 256) dist[i] = 1e10f;
    __syncthreads();

    // ---- farthest point sampling: 32 iterations ----
    int far = 0;
    for (int it = 0; it < 32; ++it) {
        const float* q0 = pcb + (size_t)far * 6;
        float cx = q0[0], cy = q0[1], cz = q0[2];
        if (tid == 0) { cenx[it] = cx; ceny[it] = cy; cenz[it] = cz; }
        float bm = -1.f; int bi = 0;
        for (int i = tid; i < NPT; i += 256) {
            const float* q = pcb + (size_t)i * 6;
            float dx = q[0] - cx, dy = q[1] - cy, dz = q[2] - cz;
            float d  = dx * dx + dy * dy + dz * dz;
            float nd = fminf(dist[i], d);
            dist[i] = nd;
            if (nd > bm) { bm = nd; bi = i; }
        }
        redv[tid] = bm; redi[tid] = bi;
        __syncthreads();
        for (int s = 128; s > 0; s >>= 1) {
            if (tid < s) {
                float v2 = redv[tid + s]; int i2 = redi[tid + s];
                if (v2 > redv[tid] || (v2 == redv[tid] && i2 < redi[tid])) {
                    redv[tid] = v2; redi[tid] = i2;
                }
            }
            __syncthreads();
        }
        far = redi[0];
        __syncthreads();
    }

    // ---- ball query: each wave handles centroids w, w+8, w+16, w+24 ----
    const float r2 = 0.02f * 0.02f;
    for (int s = w; s < 32; s += 8) {
        float cx = cenx[s], cy = ceny[s], cz = cenz[s];
        int cnt = 0;
        for (int base = 0; base < NPT && cnt < 64; base += 32) {
            int j = base + lane;
            const float* q = pcb + (size_t)j * 6;
            float dx = q[0] - cx, dy = q[1] - cy, dz = q[2] - cz;
            bool ok = (dx * dx + dy * dy + dz * dz) < r2;
            unsigned mask = __builtin_amdgcn_ballot_w32(ok);
            int pos = cnt + __builtin_popcount(mask & ((1u << lane) - 1u));
            if (ok && pos < 64) idxl[s * 64 + pos] = j;
            cnt += __builtin_popcount(mask);
        }
        if (lane == 0) cntArr[s] = cnt < 64 ? cnt : 64;
    }
    __syncthreads();

    // ---- gather: point-major rows of 9 channels (rel xyz + 6 raw pc ch) ----
    for (int e = tid; e < 32 * 64; e += 256) {
        int s = e >> 6, k = e & 63;
        int j = (k < cntArr[s]) ? idxl[e] : idxl[s * 64];
        const float* q = pcb + (size_t)j * 6;
        _Float16* row = grouped + ((size_t)b * 2048 + e) * 9;
        row[0] = (_Float16)(q[0] - cenx[s]);
        row[1] = (_Float16)(q[1] - ceny[s]);
        row[2] = (_Float16)(q[2] - cenz[s]);
        #pragma unroll
        for (int c = 0; c < 6; ++c) row[3 + c] = (_Float16)q[c];
    }
    if (tid < 96) {
        int s = tid / 3, c = tid % 3;
        float v = (c == 0) ? cenx[s] : ((c == 1) ? ceny[s] : cenz[s]);
        new_xyz[((size_t)b * 32 + s) * 3 + c] = v;
    }
}

// --------------------------- SA2 grouping ----------------------------------
// n = 32 points; nsample clipped to 32 by the reference's slice.
// feats1 point-major (b*32+j, 128); grouped2 point-major (b*1024+e, 131).
__global__ __launch_bounds__(256) void k_sa2_group(
    const float* __restrict__ xyz1,      // (B, 32, 3)
    const _Float16* __restrict__ feats1, // (B*32, 128)
    float* __restrict__ xyz2,            // (B, 32, 3)
    _Float16* __restrict__ grouped)      // (B*1024, 131)
{
    __shared__ float x1[32], y1[32], z1[32], dist[32];
    __shared__ float redv[256];
    __shared__ int   redi[256];
    __shared__ float cenx[32], ceny[32], cenz[32];
    __shared__ int   idxl[32 * 32];
    __shared__ int   cntArr[32];

    const int b = blockIdx.x;
    const int tid = threadIdx.x;
    const int lane = tid & 31, w = tid >> 5;

    if (tid < 32) {
        const float* q = xyz1 + ((size_t)b * 32 + tid) * 3;
        x1[tid] = q[0]; y1[tid] = q[1]; z1[tid] = q[2];
        dist[tid] = 1e10f;
    }
    __syncthreads();

    int far = 0;
    for (int it = 0; it < 32; ++it) {
        float cx = x1[far], cy = y1[far], cz = z1[far];
        if (tid == 0) { cenx[it] = cx; ceny[it] = cy; cenz[it] = cz; }
        float bm = -1.f; int bi = 0;
        if (tid < 32) {
            float dx = x1[tid] - cx, dy = y1[tid] - cy, dz = z1[tid] - cz;
            float d  = dx * dx + dy * dy + dz * dz;
            float nd = fminf(dist[tid], d);
            dist[tid] = nd;
            bm = nd; bi = tid;
        }
        redv[tid] = bm; redi[tid] = bi;
        __syncthreads();
        for (int s = 128; s > 0; s >>= 1) {
            if (tid < s) {
                float v2 = redv[tid + s]; int i2 = redi[tid + s];
                if (v2 > redv[tid] || (v2 == redv[tid] && i2 < redi[tid])) {
                    redv[tid] = v2; redi[tid] = i2;
                }
            }
            __syncthreads();
        }
        far = redi[0];
        __syncthreads();
    }

    const float r2 = 0.04f * 0.04f;
    for (int s = w; s < 32; s += 8) {
        float cx = cenx[s], cy = ceny[s], cz = cenz[s];
        int j = lane;
        float dx = x1[j] - cx, dy = y1[j] - cy, dz = z1[j] - cz;
        bool ok = (dx * dx + dy * dy + dz * dz) < r2;
        unsigned mask = __builtin_amdgcn_ballot_w32(ok);
        int pos = __builtin_popcount(mask & ((1u << lane) - 1u));
        if (ok) idxl[s * 32 + pos] = j;
        if (lane == 0) cntArr[s] = __builtin_popcount(mask);
    }
    __syncthreads();

    for (int e = tid; e < 32 * 32; e += 256) {
        int s = e >> 5, k = e & 31;
        int j = (k < cntArr[s]) ? idxl[e] : idxl[s * 32];
        _Float16* row = grouped + ((size_t)b * 1024 + e) * 131;
        row[0] = (_Float16)(x1[j] - cenx[s]);
        row[1] = (_Float16)(y1[j] - ceny[s]);
        row[2] = (_Float16)(z1[j] - cenz[s]);
        const _Float16* fb = feats1 + ((size_t)b * 32 + j) * 128;
        for (int c = 0; c < 128; ++c) row[3 + c] = fb[c];   // contiguous copy
    }
    if (tid < 96) {
        int s = tid / 3, c = tid % 3;
        float v = (c == 0) ? cenx[s] : ((c == 1) ? ceny[s] : cenz[s]);
        xyz2[((size_t)b * 32 + s) * 3 + c] = v;
    }
}

// --------------------------- SA3 grouping (global) --------------------------
// grouped3 point-major (b*32+p, 259); feats2 point-major (b*32+p, 256)
__global__ void k_sa3_group(
    const float* __restrict__ xyz2,
    const _Float16* __restrict__ feats2,
    _Float16* __restrict__ grouped,
    int total)
{
    int idx = blockIdx.x * 256 + threadIdx.x;
    if (idx >= total) return;
    int c = idx % 259;
    int t = idx / 259;
    int p = t % 32;
    int b = t / 32;
    _Float16 v;
    if (c < 3) v = (_Float16)xyz2[((size_t)b * 32 + p) * 3 + c];
    else       v = feats2[((size_t)b * 32 + p) * 256 + (c - 3)];
    grouped[idx] = v;
}

// ------------------------------- WMMA GEMM ----------------------------------
// Y[b,o,p] = sum_i A[o,i] * X[b*P+p, i]
//   A: (M,K) f16 row-major; X: (Bn*P, K) f16 point-major; Y: (Bn,M,P) f32
// Block = 256 threads = 8 waves, tile 32(M) x 64(P), K-step 32.
// Both operands K-contiguous -> async global->LDS staging when available.
__global__ __launch_bounds__(256) void k_gemm_f16(
    const _Float16* __restrict__ A,
    const _Float16* __restrict__ X,
    float* __restrict__ Y,
    int M, int K, int P)
{
    __shared__ _Float16 At[32 * 32];   // [m][k]
    __shared__ _Float16 Bt[64 * 32];   // [n][k]

    const int tid  = threadIdx.x;
    const int lane = tid & 31, w = tid >> 5;
    const int b     = blockIdx.z;
    const int mBase = blockIdx.y * 32;
    const int pBase = blockIdx.x * 64;
    const int mSub  = (w >> 2) << 4;
    const int nSub  = (w & 3) << 4;
    const size_t rowB = (size_t)b * P + pBase;

    const bool fullM = (mBase + 32 <= M);
    const bool fullP = (pBase + 64 <= P);
    const bool k8    = ((K & 7) == 0);      // rows 16B aligned
    const bool k4    = ((K & 3) == 0);      // rows 8B aligned

    v8f c = {};
    const int kTiles = (K + 31) >> 5;

    for (int kt = 0; kt < kTiles; ++kt) {
        const int kBase = kt << 5;
        const bool fullK = (kBase + 32 <= K);

        // ---- prefetch next K-tile rows (global_prefetch_b8) ----
        if (kt + 1 < kTiles) {
            const int kn = kBase + 32;
            if (kn < K) {
                if (tid < 32 && mBase + tid < M)
                    __builtin_prefetch(&A[(size_t)(mBase + tid) * K + kn], 0, 1);
                if (tid >= 64 && tid < 128) {
                    int p = pBase + (tid - 64);
                    if (p < P) __builtin_prefetch(&X[((size_t)b * P + p) * K + kn], 0, 1);
                }
            }
        }

#if USE_ASYNC_LDS
        if (k8 && fullM && fullP && fullK) {
            // async DMA of 16B chunks; A = 2KB (128 lanes), B = 4KB (256 lanes)
            const int rr = tid >> 2, kc = (tid & 3) << 3;
            if (tid < 128)
                async_copy_b128(A + (size_t)(mBase + rr) * K + kBase + kc,
                                &At[rr * 32 + kc]);
            async_copy_b128(X + (rowB + rr) * K + kBase + kc, &Bt[rr * 32 + kc]);
            __builtin_amdgcn_s_wait_asynccnt(0);
        } else
#endif
        {
            // ---- stage A tile [m][k] ----
            if (fullM && fullK) {
                const int m = tid >> 3, c4 = (tid & 7) << 2;
                const _Float16* src = A + (size_t)(mBase + m) * K + kBase + c4;
                if (k4) {
                    *(uint2*)&At[m * 32 + c4] = *(const uint2*)src;
                } else {
                    At[m * 32 + c4 + 0] = src[0];
                    At[m * 32 + c4 + 1] = src[1];
                    At[m * 32 + c4 + 2] = src[2];
                    At[m * 32 + c4 + 3] = src[3];
                }
            } else {
                for (int idx = tid; idx < 1024; idx += 256) {
                    int m = idx >> 5, kk = idx & 31;
                    int o = mBase + m, i = kBase + kk;
                    At[idx] = (o < M && i < K) ? A[(size_t)o * K + i] : (_Float16)0.f;
                }
            }
            // ---- stage B tile [n][k] ----
            if (fullP && fullK) {
                const int n = tid >> 2, kc = (tid & 3) << 3;
                const _Float16* src = X + (rowB + n) * K + kBase + kc;
                if (k8) {
                    *(uint4*)&Bt[n * 32 + kc] = *(const uint4*)src;
                } else {
                    #pragma unroll
                    for (int j = 0; j < 8; ++j) Bt[n * 32 + kc + j] = src[j];
                }
            } else {
                for (int idx = tid; idx < 2048; idx += 256) {
                    int n = idx >> 5, kk = idx & 31;
                    int i = kBase + kk, p = pBase + n;
                    _Float16 v = (i < K && p < P) ? X[((size_t)b * P + p) * K + i]
                                                  : (_Float16)0.f;
                    Bt[n * 32 + kk] = v;
                }
            }
        }
        __syncthreads();

        // A fragment: one contiguous 16-half run (k = 16h + s), matching B
        const int mRow = mSub + (lane & 15);
        const int off  = (lane >> 4) * 16;
        v16h a;
        const _Float16* ap0 = &At[mRow * 32 + off];
        #pragma unroll
        for (int j = 0; j < 16; ++j) a[j] = ap0[j];

        // B fragment (32x16 layout): one contiguous 16-half run
        const int nRow = nSub + (lane & 15);
        const _Float16* bp = &Bt[nRow * 32 + ((lane >> 4) << 4)];
        v16h bb;
        #pragma unroll
        for (int j = 0; j < 16; ++j) bb[j] = bp[j];

        c = __builtin_amdgcn_wmma_f32_16x16x32_f16(
                false, a, false, bb, (short)0, c, false, false);
        __syncthreads();
    }

    #pragma unroll
    for (int r = 0; r < 8; ++r) {
        int o = mBase + mSub + r + ((lane >> 4) << 3);
        int p = pBase + nSub + (lane & 15);
        if (o < M && p < P) Y[((size_t)b * M + o) * P + p] = c[r];
    }
}

// --------------------------- BN stats / apply -------------------------------
__global__ __launch_bounds__(256) void k_stats(
    const float* __restrict__ Y, const float* __restrict__ gP,
    const float* __restrict__ bP, float* __restrict__ ss,
    int Bn, int M, int P)
{
    const int o = blockIdx.x, tid = threadIdx.x;
    const size_t cnt = (size_t)Bn * P;
    double s = 0.0, s2 = 0.0;
    for (size_t idx = tid; idx < cnt; idx += 256) {
        size_t bb = idx / P, p = idx % P;
        float v = Y[((size_t)bb * M + o) * P + p];
        s += v; s2 += (double)v * v;
    }
    __shared__ double sv[256], sq[256];
    sv[tid] = s; sq[tid] = s2;
    __syncthreads();
    for (int st = 128; st > 0; st >>= 1) {
        if (tid < st) { sv[tid] += sv[tid + st]; sq[tid] += sq[tid + st]; }
        __syncthreads();
    }
    if (tid == 0) {
        double m = sv[0] / (double)cnt;
        double v = sq[0] / (double)cnt - m * m;
        float sc = gP[o] / sqrtf((float)v + EPSF);
        ss[2 * o]     = sc;
        ss[2 * o + 1] = bP[o] - (float)m * sc;
    }
}

// BN+ReLU, writing point-major f16: out[(b*P+p)*M + o]
__global__ void k_bnrelu(const float* __restrict__ Y, const float* __restrict__ ss,
                         _Float16* __restrict__ out, int M, int P, int total) {
    int idx = blockIdx.x * 256 + threadIdx.x;   // enumerates (b, p, o), o fastest
    if (idx >= total) return;
    int o = idx % M;
    int t = idx / M;
    int p = t % P;
    int b = t / P;
    float v = Y[((size_t)b * M + o) * P + p] * ss[2 * o] + ss[2 * o + 1];
    out[idx] = (_Float16)(v > 0.f ? v : 0.f);
}

// final layer: Y is (M=512, P=128 batch); d_out is (128, 512) f32
__global__ void k_bnrelu_final(const float* __restrict__ Y, const float* __restrict__ ss,
                               float* __restrict__ out, int M, int P, int total) {
    int idx = blockIdx.x * 256 + threadIdx.x;
    if (idx >= total) return;
    int o = idx / P, p = idx % P;
    float v = Y[idx] * ss[2 * o] + ss[2 * o + 1];
    out[(size_t)p * M + o] = (v > 0.f ? v : 0.f);
}

// maxpool over k: in point-major ((b*S+s)*K + k, M) -> out ((b*S+s), M)
__global__ void k_maxpool(const _Float16* __restrict__ in, _Float16* __restrict__ out,
                          int M, int K, int total) {
    int idx = blockIdx.x * 256 + threadIdx.x;   // enumerates (b, s, o), o fastest
    if (idx >= total) return;
    int o = idx % M;
    int t = idx / M;          // = b*S + s
    const _Float16* src = in + ((size_t)t * K) * M + o;
    float m = (float)src[0];
    for (int k = 1; k < K; ++k) { float v = (float)src[(size_t)k * M]; m = v > m ? v : m; }
    out[(size_t)t * M + o] = (_Float16)m;
}

// ------------------------------- host driver --------------------------------

static inline unsigned cdiv(unsigned a, unsigned b) { return (a + b - 1) / b; }

extern "C" void kernel_launch(void* const* d_in, const int* in_sizes, int n_in,
                              void* d_out, int out_size, void* d_ws, size_t ws_size,
                              hipStream_t stream) {
    (void)in_sizes; (void)n_in; (void)out_size; (void)ws_size;

    const float* pc = (const float*)d_in[0];

    static const int Ms[9] = { 64, 64, 128, 128, 128, 256, 256, 256, 512 };
    static const int Ks[9] = {  9, 64,  64, 131, 128, 128, 259, 256, 256 };
    static const int Wi[9] = {  1,  4,   7,  10,  13,  16,  19,  22,  25 };

    char* ws = (char*)d_ws;
    size_t off = 0;
    auto alloc = [&](size_t bytes) -> char* {
        size_t a = (off + 255) & ~(size_t)255;
        off = a + bytes;
        return ws + a;
    };

    _Float16* w16[11];
    for (int i = 0; i < 9; ++i) w16[i] = (_Float16*)alloc((size_t)Ms[i] * Ks[i] * 2);
    w16[9]  = (_Float16*)alloc((size_t)1024 * 512 * 2);
    w16[10] = (_Float16*)alloc((size_t)512 * 1024 * 2);

    float*     nx1    = (float*)alloc((size_t)NB * 32 * 3 * 4);
    float*     nx2    = (float*)alloc((size_t)NB * 32 * 3 * 4);
    _Float16*  feats1 = (_Float16*)alloc((size_t)NB * 32 * 128 * 2);
    _Float16*  feats2 = (_Float16*)alloc((size_t)NB * 32 * 256 * 2);
    _Float16*  feats3 = (_Float16*)alloc((size_t)NB * 512 * 2);
    _Float16*  bufA   = (_Float16*)alloc((size_t)33554432 * 2);
    _Float16*  bufB   = (_Float16*)alloc((size_t)33554432 * 2);
    float*     bufY   = (float*)alloc((size_t)33554432 * 4);
    float*     ssB    = (float*)alloc((size_t)2048 * 4);

    // ---- convert weights to f16 ----
    for (int i = 0; i < 9; ++i) {
        int n = Ms[i] * Ks[i];
        k_f32_to_f16<<<cdiv(n, 256), 256, 0, stream>>>((const float*)d_in[Wi[i]], w16[i], n);
    }
    k_f32_to_f16<<<cdiv(1024 * 512, 256), 256, 0, stream>>>((const float*)d_in[28], w16[9],  1024 * 512);
    k_f32_to_f16<<<cdiv(512 * 1024, 256), 256, 0, stream>>>((const float*)d_in[32], w16[10], 512 * 1024);

    // ---- layer runner: GEMM -> stats -> BN+ReLU (point-major out) ----
    auto run_layer = [&](const _Float16* Wh, const _Float16* Xin,
                         const float* gP, const float* bP,
                         int Bn, int M, int K, int P, _Float16* Xout) {
        dim3 g(cdiv(P, 64), cdiv(M, 32), Bn);
        k_gemm_f16<<<g, 256, 0, stream>>>(Wh, Xin, bufY, M, K, P);
        k_stats<<<M, 256, 0, stream>>>(bufY, gP, bP, ssB, Bn, M, P);
        int tot = Bn * M * P;
        k_bnrelu<<<cdiv(tot, 256), 256, 0, stream>>>(bufY, ssB, Xout, M, P, tot);
    };
    auto gpar = [&](int l) { return (const float*)d_in[Wi[l] + 1]; };
    auto bpar = [&](int l) { return (const float*)d_in[Wi[l] + 2]; };

    // ---- SA stage 1 ----
    k_sa1_group<<<NB, 256, 0, stream>>>(pc, nx1, bufA);                     // (B*2048, 9)
    run_layer(w16[0], bufA, gpar(0), bpar(0), NB,  64,   9, 2048, bufB);
    run_layer(w16[1], bufB, gpar(1), bpar(1), NB,  64,  64, 2048, bufA);
    run_layer(w16[2], bufA, gpar(2), bpar(2), NB, 128,  64, 2048, bufB);
    {
        int tot = NB * 32 * 128;   // (b,s,o)
        k_maxpool<<<cdiv(tot, 256), 256, 0, stream>>>(bufB, feats1, 128, 64, tot);
    }

    // ---- SA stage 2 ----
    k_sa2_group<<<NB, 256, 0, stream>>>(nx1, feats1, nx2, bufA);            // (B*1024, 131)
    run_layer(w16[3], bufA, gpar(3), bpar(3), NB, 128, 131, 1024, bufB);
    run_layer(w16[4], bufB, gpar(4), bpar(4), NB, 128, 128, 1024, bufA);
    run_layer(w16[5], bufA, gpar(5), bpar(5), NB, 256, 128, 1024, bufB);
    {
        int tot = NB * 32 * 256;
        k_maxpool<<<cdiv(tot, 256), 256, 0, stream>>>(bufB, feats2, 256, 32, tot);
    }

    // ---- SA stage 3 (global grouping) ----
    {
        int tot = NB * 32 * 259;
        k_sa3_group<<<cdiv(tot, 256), 256, 0, stream>>>(nx2, feats2, bufA, tot);
    }
    run_layer(w16[6], bufA, gpar(6), bpar(6), NB, 256, 259, 32, bufB);
    run_layer(w16[7], bufB, gpar(7), bpar(7), NB, 256, 256, 32, bufA);
    run_layer(w16[8], bufA, gpar(8), bpar(8), NB, 512, 256, 32, bufB);
    {
        int tot = NB * 1 * 512;    // S=1, K=32
        k_maxpool<<<cdiv(tot, 256), 256, 0, stream>>>(bufB, feats3, 512, 32, tot);
    }

    // ---- FC head (bias cancels under BN) ----
    // FC1: Y(1024,128) = W1(1024,512) x feats3(128,512); BN per feature over batch
    run_layer(w16[9], feats3, (const float*)d_in[30], (const float*)d_in[31],
              1, 1024, 512, 128, bufA);
    // FC2: Y(512,128) = W2(512,1024) x h1(128,1024)
    {
        dim3 g(cdiv(128, 64), cdiv(512, 32), 1);
        k_gemm_f16<<<g, 256, 0, stream>>>(w16[10], bufA, bufY, 512, 1024, 128);
        k_stats<<<512, 256, 0, stream>>>(bufY, (const float*)d_in[34], (const float*)d_in[35],
                                         ssB, 1, 512, 128);
        int tot = 512 * 128;
        k_bnrelu_final<<<cdiv(tot, 256), 256, 0, stream>>>(bufY, ssB, (float*)d_out,
                                                           512, 128, tot);
    }
}
